// CRATEmbedding_31155692765204
// MI455X (gfx1250) — matrix operand order repaired and checked by reading, not compiled, not run on hardware
//
#include <hip/hip_runtime.h>
#include <hip/hip_bf16.h>
#include <math.h>

#define PI_F 3.14159265358979323846f

typedef __bf16 bf16_t;
typedef __attribute__((ext_vector_type(16))) __bf16 v16bf;
typedef __attribute__((ext_vector_type(8)))  float  v8f;
typedef __attribute__((ext_vector_type(4)))  float  f32x4;

union FragAB { v16bf v; f32x4 q[2]; };

// ---------------------------------------------------------------------------
// Basis: rb[e][k] = sqrt(2/rc) * sin((k+1)*pi/rc * r) / r * switch[e]
// ---------------------------------------------------------------------------
__global__ void k_bessel(const float* __restrict__ r, const float* __restrict__ sw,
                         float* __restrict__ rb, int n, float rc) {
  int e = blockIdx.x * blockDim.x + threadIdx.x;
  if (e >= n) return;
  float x = r[e];
  float pref = sqrtf(2.0f / rc) * sw[e] / x;
#pragma unroll
  for (int k = 1; k <= 8; ++k)
    rb[e * 8 + (k - 1)] = pref * __sinf((float)k * (PI_F / rc) * x);
}

// xi0 = species_table[species]; also write zero-padded bf16 A (Mpad x 32)
__global__ void k_gather_xi0(const int* __restrict__ species, const float* __restrict__ table,
                             float* __restrict__ xi0, bf16_t* __restrict__ A, int N) {
  int i = blockIdx.x, c = threadIdx.x;  // blockDim = 32
  float v = 0.f;
  if (i < N && c < 16) {
    v = table[species[i] * 16 + c];
    xi0[i * 16 + c] = v;
  }
  A[(size_t)i * 32 + c] = (bf16_t)v;
}

// Bt[n][k] = W[k][n] (bf16, zero-padded in K). grid = Ncols, blockDim = Kpad.
__global__ void k_w_to_bt(const float* __restrict__ W, bf16_t* __restrict__ Bt,
                          int K, int Ncols) {
  int n = blockIdx.x, k = threadIdx.x, Kpad = blockDim.x;
  float v = (k < K) ? W[k * Ncols + n] : 0.f;
  Bt[(size_t)n * Kpad + k] = (bf16_t)v;
}

// Row-padded f32 -> bf16 convert. grid = Mpad, blockDim = C.
__global__ void k_f32_to_bf16_rows(const float* __restrict__ X, bf16_t* __restrict__ A,
                                   int N, int C) {
  int i = blockIdx.x, c = threadIdx.x;
  float v = (i < N) ? X[(size_t)i * C + c] : 0.f;
  A[(size_t)i * C + c] = (bf16_t)v;
}

// One wave (32 lanes) per edge: lane c handles column c of si_dst (s[:,64:96]);
// 8 atomic f32 adds per lane into mi[src]; 32 lanes cover 32 consecutive
// floats per k-slice (coalesced atomic bursts, L2-resident target).
__global__ void k_edge_scatter(const int* __restrict__ esrc, const int* __restrict__ edst,
                               const float* __restrict__ rb, const float* __restrict__ s,
                               float* __restrict__ mi, int E) {
  int e = blockIdx.x * 8 + (threadIdx.x >> 5);
  int c = threadIdx.x & 31;
  if (e >= E) return;
  int src = esrc[e], dst = edst[e];
  float sd = s[(size_t)dst * 96 + 64 + c];
  float* out = mi + (size_t)src * 256 + c;
#pragma unroll
  for (int k = 0; k < 8; ++k)
    atomicAdd(out + k * 32, rb[(size_t)e * 8 + k] * sd);
}

// da[j] = rba[j] @ W_da (8x8)
__global__ void k_da(const float* __restrict__ rba, const float* __restrict__ Wda,
                     float* __restrict__ da, int EA) {
  int j = blockIdx.x * blockDim.x + threadIdx.x;
  if (j >= EA) return;
  float rb8[8];
#pragma unroll
  for (int b = 0; b < 8; ++b) rb8[b] = rba[(size_t)j * 8 + b];
#pragma unroll
  for (int n = 0; n < 8; ++n) {
    float acc = 0.f;
#pragma unroll
    for (int b = 0; b < 8; ++b) acc += rb8[b] * Wda[b * 8 + n];
    da[(size_t)j * 8 + n] = acc;
  }
}

// 8 lanes per triplet: lane j handles channel j; the 8 lanes of one triplet
// write 8 consecutive ami addresses per harmonic -> coalesced atomic bursts.
__global__ void k_triplet_scatter(const float* __restrict__ angles, const int* __restrict__ asrc,
                                  const int* __restrict__ adst, const int* __restrict__ ca,
                                  const float* __restrict__ da, float* __restrict__ ami, int T) {
  int idx = blockIdx.x * blockDim.x + threadIdx.x;
  int t = idx >> 3;
  int j = idx & 7;
  if (t >= T) return;
  float ang = angles[t];
  int s = asrc[t], d = adst[t], c = ca[t];
  float dij = da[(size_t)s * 8 + j] * da[(size_t)d * 8 + j];
  float* out = ami + (size_t)c * 40 + j;
#pragma unroll
  for (int n = 0; n <= 4; ++n)
    atomicAdd(out + n * 8, __cosf((float)n * ang) * dij);
}

// ei0 = [xi0(16) | s[:, :64] | mi(256) | ami(40) | pad(8)] -> bf16, Kpad=384
__global__ void k_build_ei0(const float* __restrict__ xi0, const float* __restrict__ s,
                            const float* __restrict__ mi, const float* __restrict__ ami,
                            bf16_t* __restrict__ A, int N) {
  int i = blockIdx.x, c = threadIdx.x;  // blockDim = 384
  float v = 0.f;
  if (i < N) {
    if (c < 16)       v = xi0[(size_t)i * 16 + c];
    else if (c < 80)  v = s[(size_t)i * 96 + (c - 16)];
    else if (c < 336) v = mi[(size_t)i * 256 + (c - 80)];
    else if (c < 376) v = ami[(size_t)i * 40 + (c - 336)];
  }
  A[(size_t)i * 384 + c] = (bf16_t)v;
}

// ei1 = [xi1(256) | s[:, :64] | mi(256) | ami(40) | pad(24)] -> bf16, Kpad=640
__global__ void k_build_ei1(const float* __restrict__ xi1, const float* __restrict__ s,
                            const float* __restrict__ mi, const float* __restrict__ ami,
                            bf16_t* __restrict__ A, int N) {
  int i = blockIdx.x, c = threadIdx.x;  // blockDim = 640
  float v = 0.f;
  if (i < N) {
    if (c < 256)      v = xi1[(size_t)i * 256 + c];
    else if (c < 320) v = s[(size_t)i * 96 + (c - 256)];
    else if (c < 576) v = mi[(size_t)i * 256 + (c - 320)];
    else if (c < 616) v = ami[(size_t)i * 40 + (c - 576)];
  }
  A[(size_t)i * 640 + c] = (bf16_t)v;
}

// ---------------------------------------------------------------------------
// WMMA GEMM: C[M x Ncols] (f32) = A[Mpad x Kpad] (bf16) @ Bt[Ncols x Kpad]^T (bf16)
// Register-blocked: each wave owns a 16 x (NT*16) strip; A fragment is loaded
// once per K-step and fed to NT v_wmma_f32_16x16x32_bf16 ops on independent
// accumulators (hides XDL latency, loads/WMMA = (2+2NT)/NT).
// A-frag layout (16-bit A 16x32): lanes 0-15 -> K {0-7,16-23}; lanes 16-31 -> {8-15,24-31}
// B-frag (32x16): lane's column = tile_n*16 + lane%16; K {0-7,16-23}/{8-15,24-31} by half-wave.
// MODE 0: plain store. MODE 1: tssr2(x + bias). MODE 2: res + tssr2(x + bias).
// ---------------------------------------------------------------------------
template <int MODE, int NT>
__global__ void k_wmma_gemm(const bf16_t* __restrict__ A, const bf16_t* __restrict__ Bt,
                            const float* __restrict__ bias, const float* __restrict__ res,
                            float* __restrict__ C, int Mreal, int Kpad, int Ncols,
                            int tilesN, int tilesM) {
  int wave = blockIdx.x * (blockDim.x >> 5) + (threadIdx.x >> 5);
  int lane = threadIdx.x & 31;
  int wcols = tilesN / NT;            // wave-columns per M row (exact division)
  int tn0 = (wave % wcols) * NT;      // first N tile of this wave
  int tm = wave / wcols;
  if (tm >= tilesM) return;           // wave-uniform: EXEC stays all-1s for WMMA
  int m = lane & 15;
  int half = lane >> 4;
  const bf16_t* ap = A + (size_t)(tm * 16 + m) * Kpad + half * 8;
  const bf16_t* bp = Bt + (size_t)(tn0 * 16 + m) * Kpad + half * 8;
  v8f acc[NT] = {};
  for (int k = 0; k < Kpad; k += 32) {
    FragAB a;
    a.q[0] = *(const f32x4*)(ap + k);
    a.q[1] = *(const f32x4*)(ap + k + 16);
#pragma unroll
    for (int t = 0; t < NT; ++t) {
      FragAB b;
      const bf16_t* bpt = bp + (size_t)t * 16 * Kpad;
      b.q[0] = *(const f32x4*)(bpt + k);
      b.q[1] = *(const f32x4*)(bpt + k + 16);
      acc[t] = __builtin_amdgcn_wmma_f32_16x16x32_bf16(
          /*neg_a=*/false, a.v, /*neg_b=*/false, b.v,
          /*c_mod=*/(short)0, acc[t], /*reuse_a=*/false, /*reuse_b=*/false);
    }
  }
#pragma unroll
  for (int t = 0; t < NT; ++t) {
    int col = (tn0 + t) * 16 + m;
    float bcol = (MODE >= 1) ? bias[col] : 0.0f;
#pragma unroll
    for (int r = 0; r < 8; ++r) {
      int row = tm * 16 + half * 8 + r;
      if (row < Mreal) {
        float x = acc[t][r];
        if (MODE >= 1) {
          x += bcol;
          float ax = fabsf(x);
          if (ax > 1.0f) x = copysignf(2.0f * sqrtf(ax) - 1.0f, x);  // tssr2
          if (MODE == 2) x += res[(size_t)row * Ncols + col];
        }
        C[(size_t)row * Ncols + col] = x;
      }
    }
  }
}

// ---------------------------------------------------------------------------
extern "C" void kernel_launch(void* const* d_in, const int* in_sizes, int n_in,
                              void* d_out, int out_size, void* d_ws, size_t ws_size,
                              hipStream_t stream) {
  constexpr int N = 25000, E = 800000, EA = 300000, T = 1600000;
  constexpr int Mpad = 25008, tilesM = Mpad / 16;  // 1563 tiles

  const int*   species      = (const int*)d_in[0];
  const int*   edge_src     = (const int*)d_in[1];
  const int*   edge_dst     = (const int*)d_in[2];
  const float* distances    = (const float*)d_in[3];
  const float* sw           = (const float*)d_in[4];
  const float* angles       = (const float*)d_in[5];
  const int*   angle_src    = (const int*)d_in[6];
  const int*   angle_dst    = (const int*)d_in[7];
  const int*   central_atom = (const int*)d_in[8];
  const float* dist_ang     = (const float*)d_in[9];
  const float* sw_ang       = (const float*)d_in[10];
  const float* table        = (const float*)d_in[11];
  const float* W_si0        = (const float*)d_in[12];
  const float* W_si1        = (const float*)d_in[13];
  const float* W_da0        = (const float*)d_in[14];
  const float* W_da1        = (const float*)d_in[15];
  const float* W_mix0       = (const float*)d_in[16];
  const float* b_mix0       = (const float*)d_in[17];
  const float* W_mix1       = (const float*)d_in[18];
  const float* b_mix1       = (const float*)d_in[19];
  float* out = (float*)d_out;

  // workspace carve-out (~144 MB total)
  char* ws = (char*)d_ws;
  size_t off = 0;
  auto take = [&](size_t bytes) -> char* {
    char* p = ws + off;
    off += (bytes + 255) & ~(size_t)255;
    return p;
  };
  float*  rb  = (float*)take((size_t)E * 8 * 4);
  float*  rba = (float*)take((size_t)EA * 8 * 4);
  float*  da  = (float*)take((size_t)EA * 8 * 4);
  float*  xi0 = (float*)take((size_t)N * 16 * 4);
  float*  s   = (float*)take((size_t)N * 96 * 4);
  float*  mi  = (float*)take((size_t)N * 256 * 4);
  float*  ami = (float*)take((size_t)N * 40 * 4);
  float*  xi1 = (float*)take((size_t)N * 256 * 4);
  bf16_t* Abf = (bf16_t*)take((size_t)Mpad * 640 * 2);
  bf16_t* Btb = (bf16_t*)take((size_t)256 * 640 * 2);

  // GEMM launch geometry: waves = tilesM * (tilesN / NT), 8 waves / block
  constexpr int gemmS_blocks   = (tilesM * 3 + 7) / 8;  // N=96,  NT=2 -> 3 wave-cols
  constexpr int gemmMix_blocks = (tilesM * 4 + 7) / 8;  // N=256, NT=4 -> 4 wave-cols

  // radial bases (identical for both layers) + species gather
  k_bessel<<<(E + 255) / 256, 256, 0, stream>>>(distances, sw, rb, E, 5.0f);
  k_bessel<<<(EA + 255) / 256, 256, 0, stream>>>(dist_ang, sw_ang, rba, EA, 3.5f);
  k_gather_xi0<<<Mpad, 32, 0, stream>>>(species, table, xi0, Abf, N);

  // ---------------- layer 0 ----------------
  k_w_to_bt<<<96, 32, 0, stream>>>(W_si0, Btb, 16, 96);
  k_wmma_gemm<0, 2><<<gemmS_blocks, 256, 0, stream>>>(
      Abf, Btb, nullptr, nullptr, s, N, 32, 96, 6, tilesM);
  hipMemsetAsync(mi, 0, (size_t)N * 256 * 4, stream);
  hipMemsetAsync(ami, 0, (size_t)N * 40 * 4, stream);
  k_edge_scatter<<<E / 8, 256, 0, stream>>>(edge_src, edge_dst, rb, s, mi, E);
  k_da<<<(EA + 255) / 256, 256, 0, stream>>>(rba, W_da0, da, EA);
  k_triplet_scatter<<<(T * 8 + 255) / 256, 256, 0, stream>>>(
      angles, angle_src, angle_dst, central_atom, da, ami, T);
  k_build_ei0<<<Mpad, 384, 0, stream>>>(xi0, s, mi, ami, Abf, N);
  k_w_to_bt<<<256, 384, 0, stream>>>(W_mix0, Btb, 376, 256);
  k_wmma_gemm<1, 4><<<gemmMix_blocks, 256, 0, stream>>>(
      Abf, Btb, b_mix0, nullptr, xi1, N, 384, 256, 16, tilesM);

  // ---------------- layer 1 ----------------
  k_f32_to_bf16_rows<<<Mpad, 256, 0, stream>>>(xi1, Abf, N, 256);
  k_w_to_bt<<<96, 256, 0, stream>>>(W_si1, Btb, 256, 96);
  k_wmma_gemm<0, 2><<<gemmS_blocks, 256, 0, stream>>>(
      Abf, Btb, nullptr, nullptr, s, N, 256, 96, 6, tilesM);
  hipMemsetAsync(mi, 0, (size_t)N * 256 * 4, stream);
  hipMemsetAsync(ami, 0, (size_t)N * 40 * 4, stream);
  k_edge_scatter<<<E / 8, 256, 0, stream>>>(edge_src, edge_dst, rb, s, mi, E);
  k_da<<<(EA + 255) / 256, 256, 0, stream>>>(rba, W_da1, da, EA);
  k_triplet_scatter<<<(T * 8 + 255) / 256, 256, 0, stream>>>(
      angles, angle_src, angle_dst, central_atom, da, ami, T);
  k_build_ei1<<<Mpad, 640, 0, stream>>>(xi1, s, mi, ami, Abf, N);
  k_w_to_bt<<<256, 640, 0, stream>>>(W_mix1, Btb, 616, 256);
  k_wmma_gemm<2, 4><<<gemmMix_blocks, 256, 0, stream>>>(
      Abf, Btb, b_mix1, xi1, out, N, 640, 256, 16, tilesM);

  (void)in_sizes; (void)n_in; (void)out_size; (void)ws_size;
}
